// SBHopfieldSentimentPredictor_58102317580915
// MI455X (gfx1250) — compile-verified, matrix operand order checked by reading, not run on hardware
//
#include <hip/hip_runtime.h>

// ---------------- CDNA5 types ----------------
typedef __attribute__((ext_vector_type(16))) __bf16 v16bf;
typedef __attribute__((ext_vector_type(8)))  float  v8f;

union FragB16 { v16bf v; uint4 q[2]; };

#define B_   64
#define L_   256
#define V_   30000
#define E_   300
#define EPAD 320      // K for input projection, padded to mult of 32 (zero-filled)
#define H_   200
#define HPAD 224      // K for recurrent GEMM, padded to mult of 32 (zero-filled)
#define G4   800      // 4*H gate width
#define ENC_ 400
#define NH_  4
#define DH_  100

// Load one 16-lane-half fragment slice: 8 contiguous bf16 + 8 contiguous bf16 at +16.
// Matches ISA 16-bit A/B layout when caller passes base = row(k0 + half*8).
__device__ __forceinline__ v16bf load_frag(const __bf16* p) {
  FragB16 f;
  f.q[0] = *(const uint4*)(p);
  f.q[1] = *(const uint4*)(p + 16);
  return f.v;
}

__device__ __forceinline__ float sigf(float x)   { return 1.0f / (1.0f + __expf(-x)); }
__device__ __forceinline__ float tanhf_(float x) { return 1.0f - 2.0f / (__expf(2.0f * x) + 1.0f); }

// Block-wide reductions for blockDim.x == 256 (8 wave32s), result broadcast to all.
__device__ __forceinline__ float blockSum(float v, float* red) {
  #pragma unroll
  for (int o = 16; o > 0; o >>= 1) v += __shfl_xor(v, o, 32);
  __syncthreads();
  if ((threadIdx.x & 31) == 0) red[threadIdx.x >> 5] = v;
  __syncthreads();
  float s = 0.f;
  #pragma unroll
  for (int w = 0; w < 8; ++w) s += red[w];
  return s;
}
__device__ __forceinline__ float blockMax(float v, float* red) {
  #pragma unroll
  for (int o = 16; o > 0; o >>= 1) v = fmaxf(v, __shfl_xor(v, o, 32));
  __syncthreads();
  if ((threadIdx.x & 31) == 0) red[threadIdx.x >> 5] = v;
  __syncthreads();
  float s = red[0];
  #pragma unroll
  for (int w = 1; w < 8; ++w) s = fmaxf(s, red[w]);
  return s;
}

// ---------------- conversion / folding kernels ----------------
__global__ void k_conv_embed(const float* __restrict__ W, __bf16* __restrict__ out) {
  int i = blockIdx.x * 256 + threadIdx.x;          // over V_*EPAD
  if (i >= V_ * EPAD) return;
  int v = i / EPAD, k = i - v * EPAD;
  out[i] = (k < E_) ? (__bf16)W[(size_t)v * E_ + k] : (__bf16)0.0f;
}

// W:(K,800) row-major -> out:(800,Kpad) bf16, zero-padded along K
__global__ void k_conv_wT(const float* __restrict__ W, __bf16* __restrict__ out, int K, int Kpad) {
  int i = blockIdx.x * 256 + threadIdx.x;
  if (i >= G4 * Kpad) return;
  int n = i / Kpad, k = i - n * Kpad;
  out[i] = (k < K) ? (__bf16)W[k * G4 + n] : (__bf16)0.0f;
}

// wq[e,h] = sum_d Wk[e, h*DH + d] * q[h, d]   (folds Wk and q: scores = 1000 * h1 @ wq)
__global__ void k_wq(const float* __restrict__ Wk, const float* __restrict__ q, float* __restrict__ wq) {
  int i = blockIdx.x * 256 + threadIdx.x;          // over ENC_*NH_
  if (i >= ENC_ * NH_) return;
  int e = i >> 2, h = i & 3;
  float acc = 0.f;
  for (int d = 0; d < DH_; ++d) acc += Wk[e * ENC_ + h * DH_ + d] * q[h * DH_ + d];
  wq[i] = acc;
}

// ---------------- input projection: xproj[l][b][:] = emb[b,l] @ Wi + bias ----------------
// grid (128, 10), block 256 (8 waves). Wave w: M-tile blockIdx.x*8+w (16 rows of B*L),
// 5 accumulators covering 80 columns starting at blockIdx.y*80. bf16 WMMA, K = 320.
__global__ void k_proj(const int* __restrict__ x, const __bf16* __restrict__ embed_bf,
                       const __bf16* __restrict__ WiT, const float* __restrict__ bias,
                       float* __restrict__ xproj) {
  const int wave = threadIdx.x >> 5;
  const int lane = threadIdx.x & 31;
  const int hv = lane >> 4, l15 = lane & 15;
  const int m0 = (blockIdx.x * 8 + wave) * 16;
  const int n0 = blockIdx.y * 80;
  v8f acc[5] = {};
  const int tok = x[m0 + l15];
  const __bf16* arow = embed_bf + (size_t)tok * EPAD;
  for (int k0 = 0; k0 < EPAD; k0 += 32) {
    v16bf a = load_frag(arow + k0 + hv * 8);
    #pragma unroll
    for (int t = 0; t < 5; ++t) {
      v16bf b = load_frag(WiT + (size_t)(n0 + t * 16 + l15) * EPAD + k0 + hv * 8);
      acc[t] = __builtin_amdgcn_wmma_f32_16x16x32_bf16(false, a, false, b, (short)0, acc[t], false, false);
    }
  }
  #pragma unroll
  for (int t = 0; t < 5; ++t) {
    const int n = n0 + t * 16 + l15;
    const float bi = bias[n];
    #pragma unroll
    for (int r = 0; r < 8; ++r) {
      const int m = m0 + r + hv * 8;        // m = b*L + l
      const int b = m >> 8, l = m & 255;
      xproj[((size_t)l * B_ + b) * G4 + n] = acc[t][r] + bi;
    }
  }
}

// ---------------- persistent bidirectional LSTM: one workgroup per direction ----------------
// LDS: h(bf16 64x224, zero-padded cols) + c(f32 64x200) + gates(f32 64x800) = 284,672 B < 320 KB.
__global__ void __launch_bounds__(1024) k_lstm(
    const float* __restrict__ xpf, const float* __restrict__ xpb,
    const __bf16* __restrict__ whtf, const __bf16* __restrict__ whtb,
    const unsigned char* __restrict__ mask, float* __restrict__ h1) {
  extern __shared__ char smem[];
  __bf16* h_bf = (__bf16*)smem;                                               // 64 x 224
  float*  c_s  = (float*)(smem + B_ * HPAD * sizeof(__bf16));                 // 64 x 200
  float*  g_s  = (float*)(smem + B_ * HPAD * sizeof(__bf16) + B_ * H_ * sizeof(float)); // 64 x 800

  const int dir = blockIdx.x;
  const float*  xproj = dir ? xpb : xpf;
  const __bf16* WhT   = dir ? whtb : whtf;
  const int     dof   = dir ? H_ : 0;

  const int tid = threadIdx.x;
  const int wave = tid >> 5, lane = tid & 31;
  const int hv = lane >> 4, l15 = lane & 15;

  for (int i = tid; i < B_ * HPAD; i += 1024) h_bf[i] = (__bf16)0.0f;
  for (int i = tid; i < B_ * H_;   i += 1024) c_s[i] = 0.0f;
  __syncthreads();

  for (int step = 0; step < L_; ++step) {
    const int t = dir ? (L_ - 1 - step) : step;
    const float* xp = xproj + (size_t)t * B_ * G4;

    // gates = h @ Wh + xproj[t] ; 64x800 output = 4x50 WMMA tiles over 32 waves
    for (int tile = wave; tile < 200; tile += 32) {
      const int m0 = (tile / 50) * 16;
      const int n0 = (tile % 50) * 16;
      const __bf16* arow = h_bf + (m0 + l15) * HPAD;
      const __bf16* brow = WhT + (size_t)(n0 + l15) * HPAD;
      v8f acc = {};
      #pragma unroll
      for (int k0 = 0; k0 < HPAD; k0 += 32) {
        v16bf a = load_frag(arow + k0 + hv * 8);
        v16bf b = load_frag(brow + k0 + hv * 8);
        acc = __builtin_amdgcn_wmma_f32_16x16x32_bf16(false, a, false, b, (short)0, acc, false, false);
      }
      #pragma unroll
      for (int r = 0; r < 8; ++r) {
        const int m = m0 + r + hv * 8;
        g_s[m * G4 + n0 + l15] = acc[r] + xp[m * G4 + n0 + l15];
      }
    }
    __syncthreads();

    // pointwise LSTM cell + masked state update
    for (int idx = tid; idx < B_ * H_; idx += 1024) {
      const int b = idx / H_, j = idx - b * H_;
      const float* gb = g_s + b * G4;
      const float ig = gb[j], fg = gb[H_ + j], gg = gb[2 * H_ + j], og = gb[3 * H_ + j];
      const float c_old = c_s[idx];
      const float c_new = sigf(fg) * c_old + sigf(ig) * tanhf_(gg);
      const float h_new = sigf(og) * tanhf_(c_new);
      const bool  mt    = mask[b * L_ + t] != 0;
      const float h_old = (float)h_bf[b * HPAD + j];
      const float h_o   = mt ? h_new : h_old;
      c_s[idx] = mt ? c_new : c_old;
      h_bf[b * HPAD + j] = (__bf16)h_o;
      h1[((size_t)b * L_ + t) * ENC_ + dof + j] = h_o;
    }
    __syncthreads();
  }
}

// ---------------- scores + budget projection + u = p^T h1 ----------------
// grid = B*NH blocks, 256 threads, one l per thread.
__global__ void k_attn(const float* __restrict__ h1, const float* __restrict__ wq,
                       const unsigned char* __restrict__ mask,
                       float* __restrict__ p_out, float* __restrict__ u_out) {
  __shared__ float red[8];
  __shared__ float p_s[L_];
  const int bh = blockIdx.x;
  const int b = bh >> 2, h = bh & 3;
  const int l = threadIdx.x;

  const float* hrow = h1 + ((size_t)b * L_ + l) * ENC_;
  float s = 0.f;
  for (int e = 0; e < ENC_; ++e) s += hrow[e] * wq[e * 4 + h];
  s *= 1000.0f;
  const bool ml = mask[b * L_ + l] != 0;
  s = ml ? s : -1.0e9f;

  const float len    = blockSum(ml ? 1.f : 0.f, red);
  const float budget = roundf(0.2f * len);
  const float p0     = fminf(fmaxf(s, 0.f), 1.f);
  const bool  need   = blockSum(p0, red) > budget;

  float lo = 0.f;
  float hi = blockMax(ml ? s : 0.f, red) + 1.0f;
  for (int it = 0; it < 60; ++it) {
    const float mid = 0.5f * (lo + hi);
    const float val = blockSum(fminf(fmaxf(s - mid, 0.f), 1.f), red);
    const bool gt = val > budget;
    lo = gt ? mid : lo;
    hi = gt ? hi : mid;
  }
  const float tau0 = 0.5f * (lo + hi);
  const float r = s - tau0;
  const bool  fr  = (r > 0.f) && (r < 1.f) && ml;
  const bool  hi1 = (r >= 1.f) && ml;
  const float n_free = blockSum(fr ? 1.f : 0.f, red);
  const float num = blockSum(fr ? s : 0.f, red) + blockSum(hi1 ? 1.f : 0.f, red) - budget;
  float tau = num / fmaxf(n_free, 1.f);
  tau = (n_free > 0.f) ? tau : tau0;
  float p = fminf(fmaxf(s - tau, 0.f), 1.f);
  p = need ? p : p0;

  p_s[l] = p;
  p_out[bh * L_ + l] = p;
  __syncthreads();

  // u[b,h,e] = sum_l p * h1[b,l,e]
  for (int e = threadIdx.x; e < ENC_; e += 256) {
    float acc = 0.f;
    for (int ll = 0; ll < L_; ++ll)
      acc += p_s[ll] * h1[((size_t)b * L_ + ll) * ENC_ + e];
    u_out[bh * ENC_ + e] = acc;
  }
}

// ---------------- head: pooled = per-head u @ Wv ; y = sigmoid(pooled @ Wo @ out_W + out_b) ----------------
__global__ void k_final(const float* __restrict__ u, const float* __restrict__ Wv,
                        const float* __restrict__ Wo, const float* __restrict__ outW,
                        const float* __restrict__ outb, float* __restrict__ y) {
  __shared__ float pooled[ENC_];
  __shared__ float fin[ENC_];
  __shared__ float red[8];
  const int b = blockIdx.x;
  for (int i = threadIdx.x; i < ENC_; i += 256) {
    const int h = i / DH_;
    const float* ub = u + (b * NH_ + h) * ENC_;
    float acc = 0.f;
    for (int e = 0; e < ENC_; ++e) acc += ub[e] * Wv[e * ENC_ + i];
    pooled[i] = acc;
  }
  __syncthreads();
  for (int j = threadIdx.x; j < ENC_; j += 256) {
    float acc = 0.f;
    for (int i = 0; i < ENC_; ++i) acc += pooled[i] * Wo[i * ENC_ + j];
    fin[j] = acc;
  }
  __syncthreads();
  float part = 0.f;
  for (int j = threadIdx.x; j < ENC_; j += 256) part += fin[j] * outW[j];
  const float tot = blockSum(part, red);
  if (threadIdx.x == 0) y[b] = 1.0f / (1.0f + __expf(-(tot + outb[0])));
}

__global__ void k_zp(const float* __restrict__ p, const unsigned char* __restrict__ mask,
                     float* __restrict__ zp) {
  const int b = blockIdx.x, l = threadIdx.x;
  const float v = 0.25f * (p[(b * 4 + 0) * L_ + l] + p[(b * 4 + 1) * L_ + l] +
                           p[(b * 4 + 2) * L_ + l] + p[(b * 4 + 3) * L_ + l]);
  zp[b * L_ + l] = (mask[b * L_ + l] != 0) ? v : 0.f;
}

// ---------------- launch ----------------
extern "C" void kernel_launch(void* const* d_in, const int* in_sizes, int n_in,
                              void* d_out, int out_size, void* d_ws, size_t ws_size,
                              hipStream_t stream) {
  const int*           x      = (const int*)d_in[0];
  // d_in[1] = z (unused by the reference)
  const unsigned char* mask   = (const unsigned char*)d_in[2];
  const float*         embedW = (const float*)d_in[3];
  const float*         Wi_f   = (const float*)d_in[4];
  const float*         Wh_f   = (const float*)d_in[5];
  const float*         b_f    = (const float*)d_in[6];
  const float*         Wi_b   = (const float*)d_in[7];
  const float*         Wh_b   = (const float*)d_in[8];
  const float*         b_b    = (const float*)d_in[9];
  const float*         Wk     = (const float*)d_in[10];
  const float*         Wv     = (const float*)d_in[11];
  const float*         q      = (const float*)d_in[12];
  const float*         Wo     = (const float*)d_in[13];
  const float*         outW   = (const float*)d_in[14];
  const float*         outb   = (const float*)d_in[15];

  char* ws = (char*)d_ws;
  size_t off = 0;
  auto alloc = [&](size_t bytes) -> char* {
    char* p = ws + off;
    off += (bytes + 255) & ~(size_t)255;
    return p;
  };
  __bf16* embed_bf = (__bf16*)alloc((size_t)V_ * EPAD * 2);   // 19.2 MB
  __bf16* WiT_f    = (__bf16*)alloc((size_t)G4 * EPAD * 2);
  __bf16* WiT_b    = (__bf16*)alloc((size_t)G4 * EPAD * 2);
  __bf16* WhT_f    = (__bf16*)alloc((size_t)G4 * HPAD * 2);
  __bf16* WhT_b    = (__bf16*)alloc((size_t)G4 * HPAD * 2);
  float*  wq       = (float*) alloc((size_t)ENC_ * NH_ * 4);
  float*  xproj_f  = (float*) alloc((size_t)L_ * B_ * G4 * 4); // 52.4 MB, (l,b,g)
  float*  xproj_b  = (float*) alloc((size_t)L_ * B_ * G4 * 4);
  float*  h1       = (float*) alloc((size_t)B_ * L_ * ENC_ * 4); // 26.2 MB
  float*  p_ws     = (float*) alloc((size_t)B_ * NH_ * L_ * 4);
  float*  u_ws     = (float*) alloc((size_t)B_ * NH_ * ENC_ * 4);

  k_conv_embed<<<(V_ * EPAD + 255) / 256, 256, 0, stream>>>(embedW, embed_bf);
  k_conv_wT<<<(G4 * EPAD + 255) / 256, 256, 0, stream>>>(Wi_f, WiT_f, E_, EPAD);
  k_conv_wT<<<(G4 * EPAD + 255) / 256, 256, 0, stream>>>(Wi_b, WiT_b, E_, EPAD);
  k_conv_wT<<<(G4 * HPAD + 255) / 256, 256, 0, stream>>>(Wh_f, WhT_f, H_, HPAD);
  k_conv_wT<<<(G4 * HPAD + 255) / 256, 256, 0, stream>>>(Wh_b, WhT_b, H_, HPAD);
  k_wq<<<(ENC_ * NH_ + 255) / 256, 256, 0, stream>>>(Wk, q, wq);

  dim3 gp(128, 10);
  k_proj<<<gp, 256, 0, stream>>>(x, embed_bf, WiT_f, b_f, xproj_f);
  k_proj<<<gp, 256, 0, stream>>>(x, embed_bf, WiT_b, b_b, xproj_b);

  const size_t lds = (size_t)B_ * HPAD * 2 + (size_t)B_ * H_ * 4 + (size_t)B_ * G4 * 4; // 284,672 B
  k_lstm<<<2, 1024, lds, stream>>>(xproj_f, xproj_b, WhT_f, WhT_b, mask, h1);

  k_attn<<<B_ * NH_, 256, 0, stream>>>(h1, wq, mask, p_ws, u_ws);
  k_final<<<B_, 256, 0, stream>>>(u_ws, Wv, Wo, outW, outb, (float*)d_out);
  k_zp<<<B_, 256, 0, stream>>>(p_ws, mask, (float*)d_out + B_);
}
